// Attention_19602230739181
// MI455X (gfx1250) — compile-verified, matrix operand order checked by reading, not run on hardware
//
#include <hip/hip_runtime.h>
#include <hip/hip_bf16.h>

// ---------------------------------------------------------------------------
// Attention block for MI455X (gfx1250, wave32, WMMA).
// All GEMMs + QK^T + PV run on v_wmma_f32_16x16x32_bf16 (fp32 accumulate).
// GEMM: 32x64 wave tiles, register double-buffered K-loop, single moving
// pointer per matrix + static instruction offsets.
// Flash attention: block-cooperative K/V staging via CDNA5 async copies
// (global_load_async_to_lds_b128, ASYNCcnt), double buffered, uniform trip
// count + barriers; 64-wide kv tiles, online softmax in fp32; row reductions
// via DPP16 butterfly (pure VALU, no DS round-trips).
// ---------------------------------------------------------------------------

typedef __attribute__((ext_vector_type(16))) __bf16 v16bf;
typedef __attribute__((ext_vector_type(8)))  float  v8f;

#define BB   2
#define SS   2048
#define HH   2048
#define NHD  32
#define KVHD 4
#define DD   64
#define KVD  256
#define LDSLD 72            // padded LDS row stride (elements) to spread banks
#define LOG_THETA 9.2103403719761836f   // ln(10000)

union Frag16 {
    v16bf v;
    uint4 u4[2];
};

// 16x32 bf16 fragment at (lane-resolved pointer p) + constant element offset d.
// p must already include the per-lane term (lane&15)*ld + ((lane>>4)<<3),
// per CDNA5 ISA 7.12.2 16-bit A/B-matrix layout.
__device__ __forceinline__ v16bf frag_at(const __bf16* p, int d) {
    Frag16 f;
    f.u4[0] = *(const uint4*)(p + d);
    f.u4[1] = *(const uint4*)(p + d + 16);
    return f.v;
}

__device__ __forceinline__ v8f wmma_bf16(v16bf a, v16bf b, v8f c) {
    return __builtin_amdgcn_wmma_f32_16x16x32_bf16(
        false, a, false, b, (short)0, c, false, false);
}

// DPP16 butterfly reduction over each 16-lane half (matches the WMMA C-layout
// row groups). quad_perm xor1 / xor2, then half_mirror / mirror (equivalent
// to xor4 / xor8 once the smaller groups are uniform). Pure VALU - no DS.
template <int CTRL>
__device__ __forceinline__ float dpp_mov(float v) {
    return __builtin_bit_cast(float,
        __builtin_amdgcn_update_dpp(0, __builtin_bit_cast(int, v),
                                    CTRL, 0xF, 0xF, true));
}
__device__ __forceinline__ float red_max16(float v) {
    v = fmaxf(v, dpp_mov<0x0B1>(v));   // quad_perm [1,0,3,2]  (xor 1)
    v = fmaxf(v, dpp_mov<0x04E>(v));   // quad_perm [2,3,0,1]  (xor 2)
    v = fmaxf(v, dpp_mov<0x141>(v));   // row_half_mirror      (xor 4)
    v = fmaxf(v, dpp_mov<0x140>(v));   // row_mirror           (xor 8)
    return v;
}
__device__ __forceinline__ float red_sum16(float v) {
    v += dpp_mov<0x0B1>(v);
    v += dpp_mov<0x04E>(v);
    v += dpp_mov<0x141>(v);
    v += dpp_mov<0x140>(v);
    return v;
}

// CDNA5 async copy: global -> LDS, no VGPR data round-trip; tracked by
// ASYNCcnt. SADDR form: 64-bit SGPR base + 32-bit per-lane byte offset.
// lds = LDS byte address (low 32 bits of generic __shared__ pointer).
__device__ __forceinline__ void async_ld128(unsigned lds, unsigned long long sbase,
                                            unsigned voff) {
    asm volatile("global_load_async_to_lds_b128 %0, %1, %2"
                 :: "v"(lds), "v"(voff), "s"(sbase) : "memory");
}
__device__ __forceinline__ void wait_async() {
    asm volatile("s_wait_asynccnt 0x0" ::: "memory");
}

// ---------------------------------------------------------------------------
// fp32 -> bf16 elementwise convert (vectorized x4)
// ---------------------------------------------------------------------------
__global__ void k_cvt_bf16(const float* __restrict__ in, __bf16* __restrict__ out, int n4) {
    int i = blockIdx.x * blockDim.x + threadIdx.x;
    if (i >= n4) return;
    float4 f = ((const float4*)in)[i];
    __bf16 b[4] = { (__bf16)f.x, (__bf16)f.y, (__bf16)f.z, (__bf16)f.w };
    ((uint2*)out)[i] = *(uint2*)b;
}

// ---------------------------------------------------------------------------
// GEMM: C[M,N] = A[M,K] @ W[N,K]^T + bias[N].  K is a template constant so
// all row-group strides (16*K etc.) become static load offsets.
// Block = 256 threads = 8 waves: 4 waves in M x 2 waves in N.
// Each wave: 32(M) x 64(N) via 8 accumulators; K-loop step 32, register
// double-buffered so next-tile loads overlap current-tile WMMAs.
// Grid = (N/128, M/128).
// ---------------------------------------------------------------------------
template <int K>
__global__ void k_gemm(const __bf16* __restrict__ A, const __bf16* __restrict__ W,
                       const float* __restrict__ bias,
                       __bf16* __restrict__ outB, float* __restrict__ outF,
                       int N) {
    const int wave = threadIdx.x >> 5;
    const int lane = threadIdx.x & 31;
    const int wm = wave & 3;
    const int wn = wave >> 2;
    const int mbase = blockIdx.y * 128 + wm * 32;
    const int nbase = blockIdx.x * 128 + wn * 64;

    const int lrow = lane & 15;
    const int lkb  = (lane >> 4) << 3;
    const __bf16* Ap = A + (size_t)(mbase + lrow) * K + lkb;
    const __bf16* Wp = W + (size_t)(nbase + lrow) * K + lkb;

    v8f acc[8];
#pragma unroll
    for (int i = 0; i < 8; ++i) acc[i] = (v8f){};

    v16bf a0 = frag_at(Ap, 0);
    v16bf a1 = frag_at(Ap, 16 * K);
    v16bf b0 = frag_at(Wp, 0);
    v16bf b1 = frag_at(Wp, 16 * K);
    v16bf b2 = frag_at(Wp, 32 * K);
    v16bf b3 = frag_at(Wp, 48 * K);

    const __bf16* An = Ap + 32;
    const __bf16* Wn = Wp + 32;
#pragma unroll 2
    for (int k0 = 32; k0 < K; k0 += 32) {
        v16bf na0 = frag_at(An, 0);
        v16bf na1 = frag_at(An, 16 * K);
        v16bf nb0 = frag_at(Wn, 0);
        v16bf nb1 = frag_at(Wn, 16 * K);
        v16bf nb2 = frag_at(Wn, 32 * K);
        v16bf nb3 = frag_at(Wn, 48 * K);
        __builtin_prefetch(An + 32, 0, 1);     // global_prefetch_b8
        __builtin_prefetch(Wn + 32, 0, 1);

        acc[0] = wmma_bf16(a0, b0, acc[0]);
        acc[1] = wmma_bf16(a0, b1, acc[1]);
        acc[2] = wmma_bf16(a0, b2, acc[2]);
        acc[3] = wmma_bf16(a0, b3, acc[3]);
        acc[4] = wmma_bf16(a1, b0, acc[4]);
        acc[5] = wmma_bf16(a1, b1, acc[5]);
        acc[6] = wmma_bf16(a1, b2, acc[6]);
        acc[7] = wmma_bf16(a1, b3, acc[7]);

        a0 = na0; a1 = na1;
        b0 = nb0; b1 = nb1; b2 = nb2; b3 = nb3;
        An += 32; Wn += 32;
    }
    acc[0] = wmma_bf16(a0, b0, acc[0]);
    acc[1] = wmma_bf16(a0, b1, acc[1]);
    acc[2] = wmma_bf16(a0, b2, acc[2]);
    acc[3] = wmma_bf16(a0, b3, acc[3]);
    acc[4] = wmma_bf16(a1, b0, acc[4]);
    acc[5] = wmma_bf16(a1, b1, acc[5]);
    acc[6] = wmma_bf16(a1, b2, acc[6]);
    acc[7] = wmma_bf16(a1, b3, acc[7]);

    const int ncol = lane & 15;
    const int roff = (lane >> 4) << 3;
#pragma unroll
    for (int mi = 0; mi < 2; ++mi) {
#pragma unroll
        for (int r = 0; r < 8; ++r) {
            const int m = mbase + mi * 16 + r + roff;
#pragma unroll
            for (int ni = 0; ni < 4; ++ni) {
                float v = acc[mi * 4 + ni][r] + bias[nbase + ni * 16 + ncol];
                if (outF) outF[(size_t)m * N + nbase + ni * 16 + ncol] = v;
                else      outB[(size_t)m * N + nbase + ni * 16 + ncol] = (__bf16)v;
            }
        }
    }
}

// ---------------------------------------------------------------------------
// RoPE + head-major transpose (+ optional pre-scale folded into Q).
// in : [B][S][nh*64] bf16   out: [B][nh][S][64] bf16
// ---------------------------------------------------------------------------
__global__ void k_rope(const __bf16* __restrict__ in, __bf16* __restrict__ out,
                       int nh, float prescale) {
    int idx = blockIdx.x * blockDim.x + threadIdx.x;
    int total = BB * SS * nh * 32;
    if (idx >= total) return;
    int j = idx & 31;
    int h = (idx >> 5) % nh;
    int s = ((idx >> 5) / nh) % SS;
    int b = ((idx >> 5) / nh) / SS;

    float invf = __expf(-(float)j * (LOG_THETA / 32.0f));
    float ang  = (float)s * invf;
    float sn, cs;
    __sincosf(ang, &sn, &cs);

    size_t ibase = (((size_t)b * SS + s) * nh + h) * 64;
    float x0 = (float)in[ibase + j];
    float x1 = (float)in[ibase + 32 + j];

    size_t obase = (((size_t)b * nh + h) * SS + s) * 64;
    out[obase + j]      = (__bf16)((x0 * cs - x1 * sn) * prescale);
    out[obase + 32 + j] = (__bf16)((x1 * cs + x0 * sn) * prescale);
}

// ---------------------------------------------------------------------------
// Transpose V: [B][S][KVH*64] -> [B][KVH][64][S]
// ---------------------------------------------------------------------------
__global__ void k_transpose_v(const __bf16* __restrict__ in, __bf16* __restrict__ out) {
    int idx = blockIdx.x * blockDim.x + threadIdx.x;
    if (idx >= BB * SS * KVD) return;
    int d   = idx & 63;
    int kvh = (idx >> 6) & 3;
    int s   = (idx >> 8) & (SS - 1);
    int b   = idx >> 19;
    out[(((size_t)b * KVHD + kvh) * DD + d) * SS + s] = in[idx];
}

// ---------------------------------------------------------------------------
// Flash attention (causal, GQA rep=8), block-cooperative async LDS staging.
// Q: [B][NH][S][64] (pre-scaled by 1/sqrt(d)), K: [B][KVH][S][64],
// V^T: [B][KVH][64][S], Out: [B][S][H] bf16.
// Block = 256 threads = 8 waves; each wave owns 16 query rows (128/block).
// All waves iterate the block's causal kv range (uniform trip count); the
// 64x64 K and V tiles are staged with global_load_async_to_lds_b128 (no VGPR
// round-trip, ASYNCcnt), double buffered, one barrier per tile.
// Per 64-kv tile per wave: 8 score WMMAs + 8 PV WMMAs, one softmax pass.
// ---------------------------------------------------------------------------
__global__ void k_flash(const __bf16* __restrict__ Q, const __bf16* __restrict__ Kc,
                        const __bf16* __restrict__ Vt, __bf16* __restrict__ O) {
    const int tid  = threadIdx.x;
    const int wave = tid >> 5;
    const int lane = tid & 31;
    const int b = blockIdx.z;
    const int h = blockIdx.y;
    const int qb_blk = blockIdx.x * 128;
    const int qb = qb_blk + wave * 16;
    const int kvh = h >> 3;   // rep = 8

    const int lrow = lane & 15;
    const int lkb  = (lane >> 4) << 3;

    const __bf16* Qp = Q  + (((size_t)b * NHD + h) * SS + qb + lrow) * DD + lkb;
    const __bf16* Kg = Kc + (((size_t)b * KVHD + kvh) * SS) * DD;   // + kv*DD
    const __bf16* Vg = Vt + (((size_t)b * KVHD + kvh) * DD) * SS;   // + d*SS + kv

    __shared__ __align__(16) __bf16 kt[2][64 * LDSLD];
    __shared__ __align__(16) __bf16 vt[2][64 * LDSLD];
    __shared__ __align__(16) __bf16 pt[8][16 * LDSLD];

    const unsigned long long kbase = (unsigned long long)(size_t)Kg;
    const unsigned long long vbase = (unsigned long long)(size_t)Vg;

    // Cooperative async stage of one 64-kv tile into buffer bi.
    // 512 16B chunks per tile, 256 threads -> 2 chunks each per matrix.
    const int srow = tid >> 3;          // 0..31
    const int sch  = (tid & 7) << 3;    // element offset within row
    auto stage = [&](int bi, int kvb) {
#pragma unroll
        for (int i = 0; i < 2; ++i) {
            const int row = srow + (i << 5);   // 0..63
            async_ld128((unsigned)(size_t)&kt[bi][row * LDSLD + sch], kbase,
                        (unsigned)(((kvb + row) * DD + sch) * 2));
            async_ld128((unsigned)(size_t)&vt[bi][row * LDSLD + sch], vbase,
                        (unsigned)((row * SS + kvb + sch) * 2));
        }
    };

    v16bf qa0 = frag_at(Qp, 0);    // d = 0..31
    v16bf qa1 = frag_at(Qp, 32);   // d = 32..63

    v8f acc[4];
#pragma unroll
    for (int i = 0; i < 4; ++i) acc[i] = (v8f){};
    float mrow[8], lsumr[8];
#pragma unroll
    for (int r = 0; r < 8; ++r) { mrow[r] = -3.0e38f; lsumr[r] = 0.0f; }

    const __bf16* Pp = &pt[wave][0] + (size_t)lrow * LDSLD + lkb;

    const int ncol = lane & 15;
    const int roff = (lane >> 4) << 3;
    const int iters = (qb_blk + 128) / 64;   // uniform across the block

    stage(0, 0);
    wait_async();
    __syncthreads();

    int cur = 0;
    for (int it = 0; it < iters; ++it) {
        const int kvb = it * 64;
        if (it + 1 < iters) stage(cur ^ 1, (it + 1) * 64);

        const __bf16* KT = &kt[cur][0] + (size_t)lrow * LDSLD + lkb;
        const __bf16* VT = &vt[cur][0] + (size_t)lrow * LDSLD + lkb;

        // ---- scores: 4 column tiles of 16, K-dim 64 -> 8 WMMAs ----
        v8f ct[4];
#pragma unroll
        for (int t = 0; t < 4; ++t) {
            v16bf klo = frag_at(KT, t * 16 * LDSLD);
            v16bf khi = frag_at(KT, t * 16 * LDSLD + 32);
            v8f c = {};
            c = wmma_bf16(qa0, klo, c);
            c = wmma_bf16(qa1, khi, c);
            ct[t] = c;
        }

        // ---- online softmax (C-layout: m = r + 8*(lane>=16), n = lane&15) ----
#pragma unroll
        for (int r = 0; r < 8; ++r) {
            const int qg = qb + r + roff;
            float s0 = (kvb + 0  + ncol > qg) ? -3.0e38f : ct[0][r];
            float s1 = (kvb + 16 + ncol > qg) ? -3.0e38f : ct[1][r];
            float s2 = (kvb + 32 + ncol > qg) ? -3.0e38f : ct[2][r];
            float s3 = (kvb + 48 + ncol > qg) ? -3.0e38f : ct[3][r];

            float mx = red_max16(fmaxf(fmaxf(s0, s1), fmaxf(s2, s3)));
            float mnew = fmaxf(mrow[r], mx);
            float corr = __expf(mrow[r] - mnew);
            mrow[r] = mnew;

            float p0 = __expf(s0 - mnew);
            float p1 = __expf(s1 - mnew);
            float p2 = __expf(s2 - mnew);
            float p3 = __expf(s3 - mnew);
            lsumr[r] = lsumr[r] * corr + ((p0 + p1) + (p2 + p3));

            acc[0][r] *= corr; acc[1][r] *= corr; acc[2][r] *= corr; acc[3][r] *= corr;

            const int prow = r + roff;
            __bf16* pr = &pt[wave][prow * LDSLD];
            pr[ncol]      = (__bf16)p0;
            pr[16 + ncol] = (__bf16)p1;
            pr[32 + ncol] = (__bf16)p2;
            pr[48 + ncol] = (__bf16)p3;
        }

        // ---- PV: P (16x64, A-layout from LDS) x V^T tiles -> 8 WMMAs ----
        v16bf pf0 = frag_at(Pp, 0);    // kv 0..31 of tile
        v16bf pf1 = frag_at(Pp, 32);   // kv 32..63 of tile
#pragma unroll
        for (int dt = 0; dt < 4; ++dt) {
            v16bf vlo = frag_at(VT, dt * 16 * LDSLD);
            acc[dt] = wmma_bf16(pf0, vlo, acc[dt]);
        }
#pragma unroll
        for (int dt = 0; dt < 4; ++dt) {
            v16bf vhi = frag_at(VT, dt * 16 * LDSLD + 32);
            acc[dt] = wmma_bf16(pf1, vhi, acc[dt]);
        }

        wait_async();      // next buffer's async copies have landed in LDS
        __syncthreads();   // ... and everyone is done reading cur
        cur ^= 1;
    }

    // ---- finalize: divide by row sum, store [b][s][H] bf16 ----
#pragma unroll
    for (int r = 0; r < 8; ++r) {
        float lsum = red_sum16(lsumr[r]);
        float inv = 1.0f / lsum;
        const int m = qb + r + roff;
        __bf16* o = O + ((size_t)b * SS + m) * HH + h * 64 + ncol;
        o[0]  = (__bf16)(acc[0][r] * inv);
        o[16] = (__bf16)(acc[1][r] * inv);
        o[32] = (__bf16)(acc[2][r] * inv);
        o[48] = (__bf16)(acc[3][r] * inv);
    }
}

// ---------------------------------------------------------------------------
// Host-side launch
// ---------------------------------------------------------------------------
extern "C" void kernel_launch(void* const* d_in, const int* in_sizes, int n_in,
                              void* d_out, int out_size, void* d_ws, size_t ws_size,
                              hipStream_t stream) {
    const float* x  = (const float*)d_in[0];
    const float* Wq = (const float*)d_in[1];
    const float* bq = (const float*)d_in[2];
    const float* Wk = (const float*)d_in[3];
    const float* bk = (const float*)d_in[4];
    const float* Wv = (const float*)d_in[5];
    const float* bv = (const float*)d_in[6];
    const float* Wo = (const float*)d_in[7];
    const float* bo = (const float*)d_in[8];
    float* out = (float*)d_out;

    char* ws = (char*)d_ws;
    size_t off = 0;
    auto take = [&](size_t bytes) { char* p = ws + off; off += bytes; return p; };

    __bf16* xb   = (__bf16*)take((size_t)BB * SS * HH * 2);
    __bf16* Wqb  = (__bf16*)take((size_t)HH * HH * 2);
    __bf16* Wkb  = (__bf16*)take((size_t)KVD * HH * 2);
    __bf16* Wvb  = (__bf16*)take((size_t)KVD * HH * 2);
    __bf16* Wob  = (__bf16*)take((size_t)HH * HH * 2);
    __bf16* Qt   = (__bf16*)take((size_t)BB * SS * HH * 2);   // Q proj, reused as attn out
    __bf16* Kt   = (__bf16*)take((size_t)BB * SS * KVD * 2);
    __bf16* Vtmp = (__bf16*)take((size_t)BB * SS * KVD * 2);
    __bf16* Qh   = (__bf16*)take((size_t)BB * NHD * SS * DD * 2);
    __bf16* Kh   = (__bf16*)take((size_t)BB * KVHD * SS * DD * 2);
    __bf16* VT   = (__bf16*)take((size_t)BB * KVHD * DD * SS * 2);
    __bf16* Ob   = Qt;   // Qt dead after RoPE; alias for attention output

    const int M = BB * SS;   // 4096

    // 1) convert to bf16 (x4 vectorized)
    {
        int n4 = BB * SS * HH / 4;
        k_cvt_bf16<<<(n4 + 255) / 256, 256, 0, stream>>>(x, xb, n4);
        n4 = HH * HH / 4;
        k_cvt_bf16<<<(n4 + 255) / 256, 256, 0, stream>>>(Wq, Wqb, n4);
        k_cvt_bf16<<<(n4 + 255) / 256, 256, 0, stream>>>(Wo, Wob, n4);
        n4 = KVD * HH / 4;
        k_cvt_bf16<<<(n4 + 255) / 256, 256, 0, stream>>>(Wk, Wkb, n4);
        k_cvt_bf16<<<(n4 + 255) / 256, 256, 0, stream>>>(Wv, Wvb, n4);
    }

    // 2) projections (WMMA GEMMs), 128x128 block tiles, K = 2048
    k_gemm<HH><<<dim3(HH / 128,  M / 128), 256, 0, stream>>>(xb, Wqb, bq, Qt,   nullptr, HH);
    k_gemm<HH><<<dim3(KVD / 128, M / 128), 256, 0, stream>>>(xb, Wkb, bk, Kt,   nullptr, KVD);
    k_gemm<HH><<<dim3(KVD / 128, M / 128), 256, 0, stream>>>(xb, Wvb, bv, Vtmp, nullptr, KVD);

    // 3) RoPE (+ head-major transpose, Q pre-scaled by 1/sqrt(d)) and V transpose
    {
        int n = BB * SS * NHD * 32;
        k_rope<<<(n + 255) / 256, 256, 0, stream>>>(Qt, Qh, NHD, 0.125f);
        n = BB * SS * KVHD * 32;
        k_rope<<<(n + 255) / 256, 256, 0, stream>>>(Kt, Kh, KVHD, 1.0f);
        n = BB * SS * KVD;
        k_transpose_v<<<(n + 255) / 256, 256, 0, stream>>>(Vtmp, VT);
    }

    // 4) flash attention (causal, GQA, cooperative async LDS staging)
    k_flash<<<dim3(SS / 128, NHD, BB), 256, 0, stream>>>(Qh, Kh, VT, Ob);

    // 5) output projection -> fp32 d_out
    k_gemm<HH><<<dim3(HH / 128, M / 128), 256, 0, stream>>>(Ob, Wob, bo, nullptr, out, HH);
}
